// OuterAttention_11123965296644
// MI455X (gfx1250) — matrix operand order repaired and checked by reading, not compiled, not run on hardware
//
#include <hip/hip_runtime.h>
#include <hip/hip_bf16.h>
#include <math.h>

typedef __attribute__((ext_vector_type(16))) __bf16 v16bf;
typedef __attribute__((ext_vector_type(8)))  float  v8f;
typedef __attribute__((ext_vector_type(16))) unsigned short v16us;

#define A_DIM 8192
#define E_DIM 1024
#define M_DIM 1024

__device__ __forceinline__ unsigned short f2bf(float f) {
    __bf16 h = (__bf16)f;                       // fptrunc, RNE -> v_cvt_pk_bf16_f32
    return __builtin_bit_cast(unsigned short, h);
}
__device__ __forceinline__ float bf2f(unsigned short h) {
    return __uint_as_float(((unsigned int)h) << 16);
}
__device__ __forceinline__ float eluf(float x) {
    return x > 0.f ? x : expm1f(x);
}

// ---------------------------------------------------------------------------
// P[a,m] = elu( X[a,:] . W[m,:] + bias[m] + ctx[m] ), P stored as bf16.
// Block tile 128x128, 256 threads = 8 waves, each wave 64x32 (4x2 WMMA tiles).
// Software-pipelined: double-buffered LDS; raw float4 global loads for tile
// k+1 are pinned above the WMMA block by a scheduling barrier so all 8 stay
// in flight behind the WMMAs; convert+LDS-store+barrier at the bottom.
// ---------------------------------------------------------------------------
__global__ __launch_bounds__(256) void gemm_elu_kernel(
    const float* __restrict__ X, const float* __restrict__ W,
    const float* __restrict__ bias, const float* __restrict__ ctx,
    unsigned short* __restrict__ P)
{
    __shared__ unsigned short ldsA[2][128 * 32];
    __shared__ unsigned short ldsB[2][128 * 32];

    const int tid     = threadIdx.x;
    const int lane    = tid & 31;
    const int wave    = tid >> 5;
    const int laneLo  = lane & 15;
    const int laneHi  = lane >> 4;
    const int waveRow = wave & 1;   // 0..1 : 64-row halves of block tile
    const int waveCol = wave >> 1;  // 0..3 : 32-col quarters
    const int rowBase = blockIdx.x * 128;
    const int colBase = blockIdx.y * 128;

    // Staging: raw float4 registers (loads only; conversion deferred).
    float4 rA[4], rB[4];
    int stR[4], stC[4];
    #pragma unroll
    for (int i = 0; i < 4; ++i) {
        int base = (tid + 256 * i) * 4;      // element index 0..4095
        stR[i] = base >> 5;                  // row in tile
        stC[i] = base & 31;                  // col in tile (multiple of 4)
    }

    auto load_tile = [&](int k0) {
        #pragma unroll
        for (int i = 0; i < 4; ++i)
            rA[i] = *(const float4*)&X[(size_t)(rowBase + stR[i]) * E_DIM + k0 + stC[i]];
        #pragma unroll
        for (int i = 0; i < 4; ++i)
            rB[i] = *(const float4*)&W[(size_t)(colBase + stR[i]) * E_DIM + k0 + stC[i]];
    };
    auto store_tile = [&](int buf) {
        #pragma unroll
        for (int i = 0; i < 4; ++i) {
            ushort4 ha, hb;
            ha.x = f2bf(rA[i].x); ha.y = f2bf(rA[i].y);
            ha.z = f2bf(rA[i].z); ha.w = f2bf(rA[i].w);
            hb.x = f2bf(rB[i].x); hb.y = f2bf(rB[i].y);
            hb.z = f2bf(rB[i].z); hb.w = f2bf(rB[i].w);
            *(ushort4*)&ldsA[buf][stR[i] * 32 + stC[i]] = ha;
            *(ushort4*)&ldsB[buf][stR[i] * 32 + stC[i]] = hb;
        }
    };

    v8f acc[4][2] = {};

    load_tile(0);
    store_tile(0);
    __syncthreads();

    const int KT = E_DIM / 32;
    for (int kt = 0; kt < KT; ++kt) {
        const int buf = kt & 1;

        // Issue global loads for the next tile; the barrier below keeps them
        // from being sunk to their users at the loop tail, so all 8 loads
        // stay in flight while the WMMAs execute.
        if (kt + 1 < KT) load_tile((kt + 1) * 32);
        asm volatile("" ::: "memory");   // scheduling barrier: pin loads here

        // A fragments: lane<16 holds row (tile+laneLo), K={0..7,16..23};
        // lanes 16..31 hold K={8..15,24..31}   (ISA 16-bit A 16x32 layout)
        v16bf afrag[4];
        #pragma unroll
        for (int i = 0; i < 4; ++i) {
            const unsigned short* bp =
                &ldsA[buf][(waveRow * 64 + i * 16 + laneLo) * 32];
            v16us t;
            #pragma unroll
            for (int q = 0; q < 8; ++q) {
                t[q]     = bp[laneHi * 8 + q];
                t[q + 8] = bp[16 + laneHi * 8 + q];
            }
            afrag[i] = __builtin_bit_cast(v16bf, t);
        }
        // B fragments: lane holds column (tile+laneLo), K={8*laneHi..+7, 16+8*laneHi..+7}
        v16bf bfrag[2];
        #pragma unroll
        for (int j = 0; j < 2; ++j) {
            const unsigned short* bp =
                &ldsB[buf][(waveCol * 32 + j * 16 + laneLo) * 32 + laneHi * 8];
            v16us t;
            #pragma unroll
            for (int q = 0; q < 8; ++q) { t[q] = bp[q]; t[q + 8] = bp[16 + q]; }
            bfrag[j] = __builtin_bit_cast(v16bf, t);
        }

        #pragma unroll
        for (int i = 0; i < 4; ++i)
            #pragma unroll
            for (int j = 0; j < 2; ++j)
                acc[i][j] = __builtin_amdgcn_wmma_f32_16x16x32_bf16(
                    false, afrag[i], false, bfrag[j],
                    (short)0, acc[i][j], false, false);

        if (kt + 1 < KT) {
            store_tile(buf ^ 1);   // single s_wait_loadcnt lands here
            __syncthreads();       // s_wait_dscnt 0 + barrier closes WAR/RAW
        }
    }

    // Epilogue: + bias + ctx, ELU, store bf16.
    // C/D layout: lane -> column laneLo, VGPR v -> row laneHi*8 + v.
    #pragma unroll
    for (int j = 0; j < 2; ++j) {
        int m    = colBase + waveCol * 32 + j * 16 + laneLo;
        float bc = bias[m] + ctx[m];
        #pragma unroll
        for (int i = 0; i < 4; ++i) {
            int row0 = rowBase + waveRow * 64 + i * 16 + laneHi * 8;
            #pragma unroll
            for (int v = 0; v < 8; ++v) {
                float val = eluf(acc[i][j][v] + bc);
                P[(size_t)(row0 + v) * M_DIM + m] = f2bf(val);
            }
        }
    }
}

// ---------------------------------------------------------------------------
// out[r] = W[r,:] . c   (one wave per row; fp32)
// ---------------------------------------------------------------------------
__global__ __launch_bounds__(256) void matvec_rows_kernel(
    const float* __restrict__ Wm, const float* __restrict__ c,
    float* __restrict__ out, int nrows)
{
    int gw   = (blockIdx.x * 256 + threadIdx.x) >> 5;
    int lane = threadIdx.x & 31;
    if (gw >= nrows) return;
    float acc = 0.f;
    for (int k = lane; k < E_DIM; k += 32)
        acc += Wm[(size_t)gw * E_DIM + k] * c[k];
    for (int off = 16; off; off >>= 1) acc += __shfl_xor(acc, off, 32);
    if (lane == 0) out[gw] = acc;
}

// ---------------------------------------------------------------------------
// s[m] = sum_a P[a,m]   (P bf16; one thread per column, fixed order)
// ---------------------------------------------------------------------------
__global__ __launch_bounds__(256) void colsum_kernel(
    const unsigned short* __restrict__ P, float* __restrict__ s)
{
    int m = blockIdx.x * 256 + threadIdx.x;
    float acc = 0.f;
    for (int a = 0; a < A_DIM; ++a) acc += bf2f(P[(size_t)a * M_DIM + m]);
    s[m] = acc;
}

// ---------------------------------------------------------------------------
// logits[r] = P[r,:] . s   (one wave per row; P bf16)
// ---------------------------------------------------------------------------
__global__ __launch_bounds__(256) void logits_kernel(
    const unsigned short* __restrict__ P, const float* __restrict__ s,
    float* __restrict__ logits)
{
    int gw   = (blockIdx.x * 256 + threadIdx.x) >> 5;
    int lane = threadIdx.x & 31;
    float acc = 0.f;
    #pragma unroll 4
    for (int k = lane; k < M_DIM; k += 32)
        acc += bf2f(P[(size_t)gw * M_DIM + k]) * s[k];
    for (int off = 16; off; off >>= 1) acc += __shfl_xor(acc, off, 32);
    if (lane == 0) logits[gw] = acc;
}

// ---------------------------------------------------------------------------
// Softmax over 8192 logits; one 1024-thread block per array (grid.x = 2).
// ---------------------------------------------------------------------------
__global__ __launch_bounds__(1024) void softmax_kernel(
    const float* __restrict__ la, const float* __restrict__ lb,
    float* __restrict__ wa, float* __restrict__ wb)
{
    const float* x = blockIdx.x ? lb : la;
    float*       w = blockIdx.x ? wb : wa;
    __shared__ float red[1024];
    int t = threadIdx.x;

    float v[8];
    float mx = -INFINITY;
    #pragma unroll
    for (int i = 0; i < 8; ++i) { v[i] = x[t + 1024 * i]; mx = fmaxf(mx, v[i]); }
    red[t] = mx; __syncthreads();
    for (int s = 512; s > 0; s >>= 1) {
        if (t < s) red[t] = fmaxf(red[t], red[t + s]);
        __syncthreads();
    }
    mx = red[0]; __syncthreads();

    float sum = 0.f;
    #pragma unroll
    for (int i = 0; i < 8; ++i) { v[i] = __expf(v[i] - mx); sum += v[i]; }
    red[t] = sum; __syncthreads();
    for (int s = 512; s > 0; s >>= 1) {
        if (t < s) red[t] += red[t + s];
        __syncthreads();
    }
    float inv = 1.f / red[0];
    #pragma unroll
    for (int i = 0; i < 8; ++i) w[t + 1024 * i] = v[i] * inv;
}

// ---------------------------------------------------------------------------
// out[e] = sum_a seq[a,e] * w[a]   (thread per column; w staged via LDS)
// ---------------------------------------------------------------------------
__global__ __launch_bounds__(256) void att_kernel(
    const float* __restrict__ seq, const float* __restrict__ w,
    float* __restrict__ out)
{
    __shared__ float ws[256];
    int e = blockIdx.x * 256 + threadIdx.x;
    float acc = 0.f;
    for (int a0 = 0; a0 < A_DIM; a0 += 256) {
        ws[threadIdx.x] = w[a0 + threadIdx.x];
        __syncthreads();
        #pragma unroll 8
        for (int r = 0; r < 256; ++r)
            acc += seq[(size_t)(a0 + r) * E_DIM + e] * ws[r];
        __syncthreads();
    }
    out[e] = acc;
}

// ---------------------------------------------------------------------------
extern "C" void kernel_launch(void* const* d_in, const int* in_sizes, int n_in,
                              void* d_out, int out_size, void* d_ws, size_t ws_size,
                              hipStream_t stream)
{
    (void)in_sizes; (void)n_in; (void)out_size; (void)ws_size;

    const float* seq1 = (const float*)d_in[0];
    const float* seq2 = (const float*)d_in[1];
    const float* ctxc = (const float*)d_in[2];
    const float* Wc1  = (const float*)d_in[3];
    const float* Wc2  = (const float*)d_in[4];
    const float* W1   = (const float*)d_in[5];
    const float* b1   = (const float*)d_in[6];
    const float* W2   = (const float*)d_in[7];
    const float* b2   = (const float*)d_in[8];
    float* out = (float*)d_out;

    // Workspace layout (~34 MB): p1/p2 as bf16, then small f32 vectors.
    char* ws = (char*)d_ws;
    unsigned short* p1 = (unsigned short*)ws;
    unsigned short* p2 = (unsigned short*)(ws + (size_t)A_DIM * M_DIM * 2);
    float* fs   = (float*)(ws + (size_t)2 * A_DIM * M_DIM * 2);
    float* ctx1 = fs;            // 1024
    float* ctx2 = ctx1 + 1024;   // 1024
    float* s1   = ctx2 + 1024;   // 1024
    float* s2   = s1   + 1024;   // 1024
    float* la   = s2   + 1024;   // 8192
    float* lb   = la   + 8192;   // 8192
    float* wa   = lb   + 8192;   // 8192
    float* wb   = wa   + 8192;   // 8192

    // 1) ctx vectors
    matvec_rows_kernel<<<128, 256, 0, stream>>>(Wc1, ctxc, ctx1, 1024);
    matvec_rows_kernel<<<128, 256, 0, stream>>>(Wc2, ctxc, ctx2, 1024);

    // 2) fused GEMM + bias + ctx + ELU  (WMMA bf16, pipelined)
    gemm_elu_kernel<<<dim3(64, 8), 256, 0, stream>>>(seq1, W1, b1, ctx1, p1);
    gemm_elu_kernel<<<dim3(64, 8), 256, 0, stream>>>(seq2, W2, b2, ctx2, p2);

    // 3) column sums  s1 = p1.sum(0), s2 = p2.sum(0)
    colsum_kernel<<<4, 256, 0, stream>>>(p1, s1);
    colsum_kernel<<<4, 256, 0, stream>>>(p2, s2);

    // 4) logits_a = p1 @ s2 ; logits_b = p2 @ s1   (sum-factored AB^T rowsums)
    logits_kernel<<<1024, 256, 0, stream>>>(p1, s2, la);
    logits_kernel<<<1024, 256, 0, stream>>>(p2, s1, lb);

    // 5) softmaxes
    softmax_kernel<<<2, 1024, 0, stream>>>(la, lb, wa, wb);

    // 6) att1 = seq1^T @ wa ; att2 = seq2^T @ wb
    att_kernel<<<4, 256, 0, stream>>>(seq1, wa, out);
    att_kernel<<<4, 256, 0, stream>>>(seq2, wb, out + 1024);
}
